// DeepseekV3Attention_1322849927702
// MI455X (gfx1250) — matrix operand order repaired and checked by reading, not compiled, not run on hardware
//
#include <hip/hip_runtime.h>
#include <cstdint>

// Problem constants (DeepSeek-V3 MLA prefill)
#define S_  1024
#define D_  7168
#define H_  128
#define DN_ 128
#define DR_ 64
#define DV_ 128
#define DQ_ 192          // DN + DR
#define RQ_ 1536
#define RKV_ 512
#define CKV_W 576        // RKV + DR

typedef __attribute__((ext_vector_type(16))) __bf16 v16bf;
typedef __attribute__((ext_vector_type(8)))  float  v8f;

// 32-bit LDS (addrspace(3)) volatile float pointer: loads through this lower
// to real ds_load ops (not FLAT) and can never be folded away.
typedef volatile __attribute__((address_space(3))) float lds_vf;

// ---------------------------------------------------------------------------
// CDNA5 async global->LDS copy (ISA 15.18.3 op 98, tracked by ASYNCcnt).
// The LDS pointer is passed BOTH as the b32 LDS-offset operand (low 32 bits of
// the generic pointer == addrspace(3) offset) AND as a raw pointer operand so
// the LDS object escapes -- without that, LLVM proves "no stores to smA" and
// folds subsequent LDS loads to undef (observed in round 3 codegen).
// ---------------------------------------------------------------------------
static __device__ __forceinline__ void async_g2l_b128(float* lptr,
                                                      const float* gaddr) {
  asm volatile("global_load_async_to_lds_b128 %0, %1, off"
               :: "v"((unsigned)(uintptr_t)lptr), "v"(gaddr), "v"(lptr)
               : "memory");
}
static __device__ __forceinline__ void wait_asynccnt0() {
  asm volatile("s_wait_asynccnt 0x0" ::: "memory");
}

// ---------------------------------------------------------------------------
// WMMA fragment loaders (f32 memory/LDS -> bf16 fragments), contiguous k order (GEMM, QK^T).
// A (16x32, MxK): lane m=lane&15, h=lane>>4; elem e -> K=16h+e (p1 = p0 + 8)
// B (32x16, KxN): lane n=lane&15, h=lane>>4; elem e    -> K=16h+e
// ---------------------------------------------------------------------------
static __device__ __forceinline__ v16bf load_a_frag(const float* p0, const float* p1) {
  v16bf a;
#pragma unroll
  for (int e = 0; e < 8; ++e) { a[e] = (__bf16)p0[e]; a[8 + e] = (__bf16)p1[e]; }
  return a;
}

// LDS variant for data staged by the async inline asm: volatile addrspace(3)
// loads -> guaranteed ds_load, never folded, never hoisted past the waits.
static __device__ __forceinline__ v16bf load_a_frag_lds(const lds_vf* p0,
                                                        const lds_vf* p1) {
  v16bf a;
#pragma unroll
  for (int e = 0; e < 8; ++e) { a[e] = (__bf16)p0[e]; a[8 + e] = (__bf16)p1[e]; }
  return a;
}

static __device__ __forceinline__ v16bf load_b_frag(const float* p) {
  v16bf b;
#pragma unroll
  for (int e = 0; e < 16; ++e) b[e] = (__bf16)p[e];
  return b;
}

// ---------------------------------------------------------------------------
// Register-blocked GEMM: C[M,N] = A[M,K] * W[N,K]^T  (f32 in/out, bf16 WMMA)
// block = 256 threads = 8 waves arranged 2(m) x 4(n); wave tile = 32x64
// (2 M-tiles x 4 N-tiles => 8 WMMAs per 32-deep k-step).
// The 64x32 A tile is double-buffered in LDS via global_load_async_to_lds_b128
// (removes the 4x redundant A reads across the N-waves); W is read directly
// from global (b128, prefetched). grid = (M/64, ceil(N/256)).
// ---------------------------------------------------------------------------
#define MT 2
#define NT 4
__global__ void gemm_nt_wmma(const float* __restrict__ A, int lda,
                             const float* __restrict__ W,
                             float* __restrict__ C, int ldc,
                             int N, int K) {
  __shared__ float smA[2][64][32];   // 2 x 8KB, LDS offset 0
  const int tid  = threadIdx.x;
  const int lane = tid & 31;
  const int wave = tid >> 5;
  const int wm = wave >> 2;           // 0..1
  const int wn = wave & 3;            // 0..3
  const int h  = lane >> 4;
  const int lq = lane & 15;
  const int mblk = blockIdx.x * 64;
  const int n0 = blockIdx.y * 256 + wn * (NT * 16);

  const float* wrow[NT];
#pragma unroll
  for (int j = 0; j < NT; ++j) {
    int n = n0 + j * 16 + lq;
    n = (n < N) ? n : (N - 1);        // clamp OOB columns to a valid row
    wrow[j] = W + (size_t)n * (size_t)K;
  }

  v8f acc[MT][NT];
#pragma unroll
  for (int i = 0; i < MT; ++i)
#pragma unroll
    for (int j = 0; j < NT; ++j)
#pragma unroll
      for (int r = 0; r < 8; ++r) acc[i][j][r] = 0.f;

  // Stage a 64x32 f32 A tile into LDS buffer `buf`: 512 16B chunks, 2/thread.
  auto stageA = [&](int buf, int k0) {
#pragma unroll
    for (int j = 0; j < 2; ++j) {
      const int ci  = tid + 256 * j;      // chunk id, 0..511
      const int row = ci >> 3;            // 0..63
      const int col = (ci & 7) * 4;       // 0..28
      async_g2l_b128(&smA[buf][row][col],
                     A + (size_t)(mblk + row) * lda + k0 + col);
    }
  };

  stageA(0, 0);
  int buf = 0;
  for (int k0 = 0; k0 < K; k0 += 32) {
    wait_asynccnt0();       // this wave's async writes for `buf` are in LDS
    __syncthreads();        // everyone's writes done + prior step's reads done
    if (k0 + 32 < K) stageA(buf ^ 1, k0 + 32);   // overlaps compute below

    v16bf af[MT], bf[NT];
#pragma unroll
    for (int i = 0; i < MT; ++i) {
      // low 32 bits of the generic pointer == LDS byte offset (ISA aperture
      // rule); as3 pointers are 32-bit on amdgcn so the cast is exact.
      const lds_vf* ar =
          (const lds_vf*)(unsigned)(uintptr_t)&smA[buf][wm * 32 + i * 16 + lq][0];
      af[i] = load_a_frag_lds(ar + 16 * h, ar + 16 * h + 8);
    }
#pragma unroll
    for (int j = 0; j < NT; ++j) {
      bf[j] = load_b_frag(wrow[j] + k0 + 16 * h);
      __builtin_prefetch(wrow[j] + k0 + 32 + 16 * h, 0, 1);   // global_prefetch_b8
    }
#pragma unroll
    for (int i = 0; i < MT; ++i)
#pragma unroll
      for (int j = 0; j < NT; ++j)
        acc[i][j] = __builtin_amdgcn_wmma_f32_16x16x32_bf16(
            false, af[i], false, bf[j], (short)0, acc[i][j], false, false);
    buf ^= 1;
  }

#pragma unroll
  for (int i = 0; i < MT; ++i)
#pragma unroll
    for (int j = 0; j < NT; ++j) {
      if (n0 + j * 16 < N) {          // N is always a multiple of 16 here
#pragma unroll
        for (int r = 0; r < 8; ++r)
          C[(size_t)(mblk + wm * 32 + i * 16 + r + 8 * h) * ldc + n0 + j * 16 + lq] =
              acc[i][j][r];
      }
    }
}

// ---------------------------------------------------------------------------
// RMSNorm, in-place, one row per block. `ld` = row stride (>= ncols).
// ---------------------------------------------------------------------------
__global__ void rmsnorm_kernel(float* __restrict__ x, const float* __restrict__ w,
                               int ncols, int ld) {
  const int row = blockIdx.x;
  float* px = x + (size_t)row * ld;
  float ss = 0.f;
  for (int i = threadIdx.x; i < ncols; i += blockDim.x) { float v = px[i]; ss += v * v; }
  __shared__ float red[256];
  red[threadIdx.x] = ss;
  __syncthreads();
  for (int s = 128; s > 0; s >>= 1) {
    if ((int)threadIdx.x < s) red[threadIdx.x] += red[threadIdx.x + s];
    __syncthreads();
  }
  const float inv = rsqrtf(red[0] / (float)ncols + 1e-6f);
  for (int i = threadIdx.x; i < ncols; i += blockDim.x) px[i] = px[i] * inv * w[i];
}

// ---------------------------------------------------------------------------
// RoPE (rotate-half convention of the reference), applied in place.
// ---------------------------------------------------------------------------
__global__ void rope_q_kernel(float* __restrict__ qf, const float* __restrict__ c,
                              const float* __restrict__ s) {
  const int idx = blockIdx.x * blockDim.x + threadIdx.x;
  const int i    = idx & 31;
  const int head = (idx >> 5) & (H_ - 1);
  const int row  = idx >> 12;
  if (row >= S_) return;
  float* pe = qf + ((size_t)row * H_ + head) * DQ_ + DN_;
  const float c0 = c[row * DR_ + i],      c1 = c[row * DR_ + 32 + i];
  const float s0 = s[row * DR_ + i],      s1 = s[row * DR_ + 32 + i];
  const float x0 = pe[i], x1 = pe[32 + i];
  pe[i]      = x0 * c0 - x1 * s0;
  pe[32 + i] = x1 * c1 + x0 * s1;
}

__global__ void rope_k_kernel(float* __restrict__ ckv, const float* __restrict__ c,
                              const float* __restrict__ s) {
  const int idx = blockIdx.x * blockDim.x + threadIdx.x;
  const int i   = idx & 31;
  const int row = idx >> 5;
  if (row >= S_) return;
  float* pe = ckv + (size_t)row * CKV_W + RKV_;
  const float c0 = c[row * DR_ + i],      c1 = c[row * DR_ + 32 + i];
  const float s0 = s[row * DR_ + i],      s1 = s[row * DR_ + 32 + i];
  const float x0 = pe[i], x1 = pe[32 + i];
  pe[i]      = x0 * c0 - x1 * s0;
  pe[32 + i] = x1 * c1 + x0 * s1;
}

// ---------------------------------------------------------------------------
// V transpose: vt[h, d, s] = kvbuf[s, h*256 + 128 + d]
// Makes P*V B-fragment loads contiguous along the key axis (4 b128 per frag).
// ---------------------------------------------------------------------------
__global__ void transpose_v_kernel(const float* __restrict__ kvbuf,
                                   float* __restrict__ vt) {
  const int idx = blockIdx.x * blockDim.x + threadIdx.x;
  const int d  = idx & (DV_ - 1);
  const int hh = (idx >> 7) & (H_ - 1);
  const int s  = idx >> 14;
  if (s >= S_) return;
  vt[((size_t)hh * DV_ + d) * S_ + s] = kvbuf[((size_t)s * H_ + hh) * 256 + DN_ + d];
}

// ---------------------------------------------------------------------------
// MLA flash attention: one wave = one (head, 16-query block).
// Scores computed TRANSPOSED (S^T = K * Q^T) so the two 16x16 C-layout score
// tiles of a 32-key step are directly the A-fragment for the P*V WMMA
// (C layout M=r+8h striped over lanes == A layout K=8h+e / 16+8h+(e-8)).
// q:   (S, H, 192)  roped full query
// kv:  (S, H, 256)  [k_nope | v]       (k_nope side used here)
// ckv: (S, 576)     k_pe at offset 512 (roped)
// vt:  (H, 128, S)  transposed V
// o:   (S, H, 128)
// ---------------------------------------------------------------------------
__global__ void mla_attn(const float* __restrict__ q, const float* __restrict__ kv,
                         const float* __restrict__ ckv, const float* __restrict__ vt,
                         float* __restrict__ o) {
  const int lane = threadIdx.x & 31;
  const int wave = threadIdx.x >> 5;
  const int gw   = blockIdx.x * 8 + wave;
  const int qb   = gw & 63;          // S/16 = 64 query blocks
  const int head = gw >> 6;
  const int h  = lane >> 4;
  const int lq = lane & 15;
  const int q_row = qb * 16 + lq;
  const float scale = 0.07216878364870322f;  // 192^-0.5

  // Hoist Q fragments (B-frag of S^T wmma): lane=query, elems = q dims 32c+16h+e
  const float* qptr = q + ((size_t)q_row * H_ + head) * DQ_;
  v16bf qfr[6];
#pragma unroll
  for (int c = 0; c < 6; ++c) qfr[c] = load_b_frag(qptr + c * 32 + 16 * h);

  const float* vbase = vt + (size_t)head * DV_ * S_ + (size_t)lq * S_;

  v8f acc[8];
#pragma unroll
  for (int t = 0; t < 8; ++t)
#pragma unroll
    for (int i = 0; i < 8; ++i) acc[t][i] = 0.f;

  float mrun = -1e30f;
  float lrun = 0.f;
  const int kend = qb * 16 + 16;  // exclusive causal bound for this block

  for (int kb = 0; kb < kend; kb += 32) {
    // ---- transposed score tiles: keys kb..kb+15 (t=0), kb+16..kb+31 (t=1)
    v8f st[2];
#pragma unroll
    for (int t = 0; t < 2; ++t)
#pragma unroll
      for (int i = 0; i < 8; ++i) st[t][i] = 0.f;

#pragma unroll
    for (int t = 0; t < 2; ++t) {
      if (kb + 16 * t < kend) {  // wave-uniform
        const int key = kb + 16 * t + lq;  // A-frag row = key
        const float* kn = kv + ((size_t)key * H_ + head) * 256;     // dims [0,128)
        const float* kp = ckv + (size_t)key * CKV_W + (RKV_ - DN_); // kp[d], d in [128,192)
#pragma unroll
        for (int c = 0; c < 6; ++c) {
          const int d0 = c * 32 + 16 * h;  // 8-aligned; never straddles the 128 boundary
          const int d1 = d0 + 8;
          const float* p0 = ((d0 < DN_) ? kn : kp) + d0;
          const float* p1 = ((d1 < DN_) ? kn : kp) + d1;
          v16bf a = load_a_frag(p0, p1);
          st[t] = __builtin_amdgcn_wmma_f32_16x16x32_bf16(false, a, false, qfr[c],
                                                          (short)0, st[t], false, false);
        }
      }
    }

    // ---- scale + causal mask + online softmax (query = lane&15, halves merged)
    float colmax = -1e30f;
#pragma unroll
    for (int t = 0; t < 2; ++t)
#pragma unroll
      for (int r = 0; r < 8; ++r) {
        const int key_idx = kb + 16 * t + r + 8 * h;
        float v = st[t][r] * scale;
        v = (key_idx <= q_row) ? v : -1e30f;
        st[t][r] = v;
        colmax = fmaxf(colmax, v);
      }
    colmax = fmaxf(colmax, __shfl_xor(colmax, 16));
    const float mnew = fmaxf(mrun, colmax);
    const float corr = __expf(mrun - mnew);
    float psum = 0.f;
    v16bf pf;  // A-fragment of P for the P*V WMMA, built with no shuffles
#pragma unroll
    for (int t = 0; t < 2; ++t)
#pragma unroll
      for (int r = 0; r < 8; ++r) {
        const float pv = __expf(st[t][r] - mnew);
        psum += pv;
        pf[t * 8 + r] = (__bf16)pv;
      }
    psum += __shfl_xor(psum, 16);
    lrun = lrun * corr + psum;
    mrun = mnew;

    // rescale accumulator rows: row M=r+8h needs corr of query (r+8h)
#pragma unroll
    for (int r = 0; r < 8; ++r) {
      const float cr = __shfl(corr, r + 8 * h);
#pragma unroll
      for (int t = 0; t < 8; ++t) acc[t][r] *= cr;
    }

    // ---- P (16x32) * V (32x128): 8 WMMA tiles along DV, contiguous V loads.
    // V keys kb+8h+e / kb+16+8h+e (pf's k order), all < S; masked keys'
    // probabilities are exactly 0, so loaded values are don't-cares.
    const int ks = kb + 8 * h;
#pragma unroll
    for (int t = 0; t < 8; ++t) {
      const float* vp = vbase + (size_t)(t * 16) * S_ + ks;
      v16bf bv = load_a_frag(vp, vp + 16);
      acc[t] = __builtin_amdgcn_wmma_f32_16x16x32_bf16(false, pf, false, bv, (short)0,
                                                       acc[t], false, false);
    }
  }

  // ---- normalize and store (C layout: row = qb*16 + r + 8h, col = t*16 + lq)
  const float linv = 1.f / lrun;
#pragma unroll
  for (int r = 0; r < 8; ++r) {
    const float li = __shfl(linv, r + 8 * h);
    const int row = qb * 16 + r + 8 * h;
#pragma unroll
    for (int t = 0; t < 8; ++t)
      o[((size_t)row * H_ + head) * DV_ + t * 16 + lq] = acc[t][r] * li;
  }
}

// ---------------------------------------------------------------------------
extern "C" void kernel_launch(void* const* d_in, const int* in_sizes, int n_in,
                              void* d_out, int out_size, void* d_ws, size_t ws_size,
                              hipStream_t stream) {
  (void)in_sizes; (void)n_in; (void)out_size; (void)ws_size;
  const float* hs    = (const float*)d_in[0];
  const float* cosb  = (const float*)d_in[1];
  const float* sinb  = (const float*)d_in[2];
  const float* wq_a  = (const float*)d_in[3];
  const float* q_ln  = (const float*)d_in[4];
  const float* wq_b  = (const float*)d_in[5];
  const float* wkv_a = (const float*)d_in[6];
  const float* kv_ln = (const float*)d_in[7];
  const float* wkv_b = (const float*)d_in[8];
  const float* wo    = (const float*)d_in[9];
  float* out = (float*)d_out;

  float* p     = (float*)d_ws;
  float* q_lat = p; p += (size_t)S_ * RQ_;          // (S, 1536)
  float* ckv   = p; p += (size_t)S_ * CKV_W;        // (S, 576)
  float* qfull = p; p += (size_t)S_ * H_ * DQ_;     // (S, H, 192)
  float* kvbuf = p; p += (size_t)S_ * H_ * 256;     // (S, H, 256)
  float* attno = p; p += (size_t)S_ * H_ * DV_;     // (S, H, 128)
  float* vt    = p; p += (size_t)H_ * DV_ * S_;     // (H, 128, S)

  // 1) low-rank projections from hidden states
  gemm_nt_wmma<<<dim3(S_ / 64, (RQ_ + 255) / 256), 256, 0, stream>>>(
      hs, D_, wq_a, q_lat, RQ_, RQ_, D_);
  gemm_nt_wmma<<<dim3(S_ / 64, (CKV_W + 255) / 256), 256, 0, stream>>>(
      hs, D_, wkv_a, ckv, CKV_W, CKV_W, D_);

  // 2) RMSNorm (kv norm touches only the first 512 of each 576-wide row)
  rmsnorm_kernel<<<S_, 256, 0, stream>>>(q_lat, q_ln, RQ_, RQ_);
  rmsnorm_kernel<<<S_, 256, 0, stream>>>(ckv, kv_ln, RKV_, CKV_W);

  // 3) rope k_pe (in place inside ckv)
  rope_k_kernel<<<(S_ * 32) / 256, 256, 0, stream>>>(ckv, cosb, sinb);

  // 4) up-projections
  gemm_nt_wmma<<<dim3(S_ / 64, (H_ * DQ_) / 256), 256, 0, stream>>>(
      q_lat, RQ_, wq_b, qfull, H_ * DQ_, H_ * DQ_, RQ_);
  gemm_nt_wmma<<<dim3(S_ / 64, (H_ * 256) / 256), 256, 0, stream>>>(
      ckv, CKV_W, wkv_b, kvbuf, H_ * 256, H_ * 256, RKV_);

  // 5) rope q_pe (in place) + V transpose
  rope_q_kernel<<<(S_ * H_ * 32) / 256, 256, 0, stream>>>(qfull, cosb, sinb);
  transpose_v_kernel<<<(S_ * H_ * DV_) / 256, 256, 0, stream>>>(kvbuf, vt);

  // 6) causal flash attention, WMMA for both QK^T and P*V
  mla_attn<<<(S_ / 16) * H_ / 8, 256, 0, stream>>>(qfull, kvbuf, ckv, vt, attno);

  // 7) output projection into d_out
  gemm_nt_wmma<<<dim3(S_ / 64, D_ / 256), 256, 0, stream>>>(
      attno, H_ * DV_, wo, out, D_, D_, H_ * DV_);
}
